// MultiHeadAttention_77567109365891
// MI455X (gfx1250) — matrix operand order repaired and checked by reading, not compiled, not run on hardware
//
#include <hip/hip_runtime.h>

// MHA (B=1, S=2048, E=2048, H=16, D=128) for gfx1250 (MI455X), wave32.
// Pipeline: f32->bf16 convert; QKV projections via v_wmma_f32_16x16x32_bf16
// with 64-wide W tiles double-buffered in LDS via global_load_async_to_lds_b128
// and fence-free split barriers; RoPE; flash attention (S^T = K*Q^T,
// O^T = V^T * P^T, softmax per-lane); output projection to f32.

#define EMBED 2048
#define SEQ   2048
#define NHEAD 16
#define HDIM  128
#define NELEM (EMBED * SEQ)

typedef __attribute__((ext_vector_type(16))) __bf16 v16bf;
typedef __attribute__((ext_vector_type(8)))  __bf16 v8bf;
typedef __attribute__((ext_vector_type(4)))  __bf16 v4bf;
typedef __attribute__((ext_vector_type(8)))  float  v8f;
typedef __attribute__((ext_vector_type(4)))  int    v4i;

typedef __attribute__((address_space(1))) v4i* glob_b128_t;
typedef __attribute__((address_space(3))) v4i* lds_b128_t;

__device__ __forceinline__ v8f wmma_bf16(v16bf a, v16bf b, v8f c) {
  // D = A(16x32 bf16) * B(32x16 bf16) + C(16x16 f32)
  return __builtin_amdgcn_wmma_f32_16x16x32_bf16(false, a, false, b, (short)0, c,
                                                 false, false);
}

__device__ __forceinline__ v16bf join8(v8bf lo, v8bf hi) {
  v16bf r;
#pragma unroll
  for (int i = 0; i < 8; ++i) { r[i] = lo[i]; r[8 + i] = hi[i]; }
  return r;
}

// A fragment 16(M) x 32(K), 16-bit elements:
//  lane<16 : M=lane,    elems e -> K=k0+e     (k0+0..15)
//  lane>=16: M=lane-16, elems e -> K=k0+16+e  (k0+16..31)
__device__ __forceinline__ v16bf load_A(const __bf16* base, int ld, int row0,
                                        int k0, int lane) {
  const int m  = lane & 15;
  const int ko = (lane < 16) ? 0 : 16;
  const __bf16* p = base + (size_t)(row0 + m) * ld + (k0 + ko);
  v8bf lo = *(const v8bf*)p;         // 16B -> global_load_b128 / ds_load_b128
  v8bf hi = *(const v8bf*)(p + 8);
  return join8(lo, hi);
}

// B fragment 32(K) x 16(N) where B[k][n] = src[n0+n][k0+k] (rows of src are
// K-contiguous):
//  lane<16 : N=lane,    elems e -> K=e       (k0+0..15)
//  lane>=16: N=lane-16, elems e -> K=16+e    (k0+16..31)
__device__ __forceinline__ v16bf load_BrowT(const __bf16* base, int ld, int n0,
                                            int k0, int lane) {
  const int n  = lane & 15;
  const int ko = (lane < 16) ? 0 : 16;
  const __bf16* p = base + (size_t)(n0 + n) * ld + (k0 + ko);
  v8bf lo = *(const v8bf*)p;
  v8bf hi = *(const v8bf*)(p + 8);
  return join8(lo, hi);
}

// ---------------------------------------------------------------- convert
__global__ __launch_bounds__(256) void cvt_f32_to_bf16(
    const float* __restrict__ src, __bf16* __restrict__ dst) {
  size_t i = ((size_t)blockIdx.x * blockDim.x + threadIdx.x) * 4;
  float4 f = *(const float4*)(src + i);
  v4bf b;
  b[0] = (__bf16)f.x; b[1] = (__bf16)f.y; b[2] = (__bf16)f.z; b[3] = (__bf16)f.w;
  *(v4bf*)(dst + i) = b;
}

// ---------------------------------------------------------------- GEMM
// C[s, f] = sum_e A[s, e] * W[f, e]   (A, W bf16 2048x2048 row-major)
// W tiles (64 f x 64 e = 8 KB) staged in LDS with async copies, double
// buffered; the 8 waves of the block share one copy instead of 8 global
// reads. Split barriers (no fence) keep the prefetched A global loads in
// flight across the sync: LDS visibility is already guaranteed by each
// wave's s_wait_asynccnt 0 before it signals.
// mode 0: write bf16 row-major [s][f]     (raw Q, K)
// mode 1: write bf16 transposed [f][s]    (V^T, so attention A-frags are contiguous)
// mode 2: write f32 row-major [s][f]      (final output)
__global__ __launch_bounds__(256) void gemm_xwt(
    const __bf16* __restrict__ A, const __bf16* __restrict__ W,
    __bf16* outb, float* outf, int mode) {
  __shared__ __bf16 lw[2][64 * 64];   // 2 x 8 KB W tiles
  const int tid  = threadIdx.x;
  const int lane = tid & 31;
  const int wv   = tid >> 5;
  const int row0 = blockIdx.y * 128 + wv * 16;  // s tile
  const int col0 = blockIdx.x * 64;             // f tile (4 x 16)

  // Thread tid copies 2x16 bytes of W row col0 + tid/4 (64-elem row).
  auto issue = [&](int buf, int e0) {
    const __bf16* g = W + (size_t)(col0 + (tid >> 2)) * EMBED + e0 + (tid & 3) * 8;
    __bf16* l = &lw[buf][(tid >> 2) * 64 + (tid & 3) * 8];
    __builtin_amdgcn_global_load_async_to_lds_b128(
        (glob_b128_t)g, (lds_b128_t)l, 0, 0);
    __builtin_amdgcn_global_load_async_to_lds_b128(
        (glob_b128_t)(g + 32), (lds_b128_t)(l + 32), 0, 0);
  };

  issue(0, 0);
  v8f acc[4] = {};
  v16bf a0 = load_A(A, EMBED, row0,  0, lane);  // prefetched A fragments
  v16bf a1 = load_A(A, EMBED, row0, 32, lane);
  int it = 0;
  for (int e0 = 0; e0 < EMBED; e0 += 64, ++it) {
    asm volatile("s_wait_asynccnt 0x0" ::: "memory");  // my async writes landed
    __builtin_amdgcn_s_barrier_signal(-1);   // fence-free workgroup barrier:
    __builtin_amdgcn_s_barrier_wait(-1);     // everyone's tile data is in LDS
    const bool more = (e0 + 64 < EMBED);
    if (more) issue((it + 1) & 1, e0 + 64);
    const __bf16* wb = lw[it & 1];
    v16bf an0 = more ? load_A(A, EMBED, row0, e0 + 64, lane) : a0;
    v16bf an1 = more ? load_A(A, EMBED, row0, e0 + 96, lane) : a1;
#pragma unroll
    for (int t = 0; t < 4; ++t) {
      v16bf b0 = load_BrowT(wb, 64, 16 * t,  0, lane);  // ds_load_b128 x2
      v16bf b1 = load_BrowT(wb, 64, 16 * t, 32, lane);
      acc[t] = wmma_bf16(a0, b0, acc[t]);
      acc[t] = wmma_bf16(a1, b1, acc[t]);
    }
    a0 = an0; a1 = an1;
  }

  // C/D layout: lane<16: N=lane, M=j ; lane>=16: N=lane-16, M=8+j
  const int n  = lane & 15;
  const int mh = (lane < 16) ? 0 : 8;
#pragma unroll
  for (int t = 0; t < 4; ++t) {
    const int c = col0 + 16 * t + n;
    if (mode == 0) {
#pragma unroll
      for (int j = 0; j < 8; ++j)
        outb[(size_t)(row0 + mh + j) * EMBED + c] = (__bf16)acc[t][j];
    } else if (mode == 1) {
      v8bf p;
#pragma unroll
      for (int j = 0; j < 8; ++j) p[j] = (__bf16)acc[t][j];
      *(v8bf*)(outb + (size_t)c * SEQ + row0 + mh) = p;  // V^T[f][s]
    } else {
#pragma unroll
      for (int j = 0; j < 8; ++j)
        outf[(size_t)(row0 + mh + j) * EMBED + c] = acc[t][j];
    }
  }
}

// ---------------------------------------------------------------- RoPE
// q' = q*cos + rotate_half(q)*sin ; one thread per (s, h, d2<64), in place.
__global__ __launch_bounds__(256) void rotary_inplace(
    __bf16* Q, __bf16* K, const float* __restrict__ cosb,
    const float* __restrict__ sinb) {
  const int idx = blockIdx.x * blockDim.x + threadIdx.x;
  const int d2 = idx & 63;
  const int h  = (idx >> 6) & 15;
  const int s  = idx >> 10;
  const size_t r = (size_t)s * EMBED + h * HDIM + d2;
  const float c0 = cosb[s * HDIM + d2], c1 = cosb[s * HDIM + d2 + 64];
  const float s0 = sinb[s * HDIM + d2], s1 = sinb[s * HDIM + d2 + 64];
  float q0 = (float)Q[r], q1 = (float)Q[r + 64];
  Q[r]      = (__bf16)(q0 * c0 - q1 * s0);
  Q[r + 64] = (__bf16)(q1 * c1 + q0 * s1);
  float k0 = (float)K[r], k1 = (float)K[r + 64];
  K[r]      = (__bf16)(k0 * c0 - k1 * s0);
  K[r + 64] = (__bf16)(k1 * c1 + k0 * s1);
}

// ---------------------------------------------------------------- flash attn
// One wave per (head, 16-query tile). 32 keys per iteration.
// S^T = K * Q^T  (so a softmax row q is per-lane: lane&15 == q)
// O^T = V^T * P^T (Ot accumulators: lane&15 == q, M == d)
__global__ __launch_bounds__(128) void flash_attn(
    const __bf16* __restrict__ Qr, const __bf16* __restrict__ Kr,
    const __bf16* __restrict__ Vt, __bf16* __restrict__ Ob) {
  const int lane = threadIdx.x & 31;
  const int wv   = threadIdx.x >> 5;
  const int h    = blockIdx.x;
  const int qt   = blockIdx.y * 4 + wv;
  const int q0   = qt * 16;
  const int hq   = lane & 15;               // query column for this lane
  const int kloc = (lane < 16) ? 0 : 8;     // k_local base in S^T tiles
  const float SCL = 0.08838834764831845f * 1.4426950408889634f; // 1/sqrt(128)*log2(e)

  // Q^T B-fragments (loop invariant): B[d][q] = Qr[q0+q][h*128 + d]
  v16bf Bq[4];
#pragma unroll
  for (int c = 0; c < 4; ++c)
    Bq[c] = load_BrowT(Qr, EMBED, q0, h * HDIM + 32 * c, lane);

  v8f accO[8] = {};                 // O^T: 8 tiles of (16 d x 16 q)
  float m = -1e30f, l = 0.0f;

  const int nk = qt / 2 + 1;        // number of 32-key chunks (causal)
  for (int ck = 0; ck < nk; ++ck) {
    const int kb = ck * 32;

    // Batch all 8 K A-fragment loads, then the 8 score WMMAs.
    v16bf ak[8];
#pragma unroll
    for (int c = 0; c < 4; ++c) {
      ak[2 * c]     = load_A(Kr, EMBED, kb,      h * HDIM + 32 * c, lane);
      ak[2 * c + 1] = load_A(Kr, EMBED, kb + 16, h * HDIM + 32 * c, lane);
    }
    v8f st0 = {}, st1 = {};
#pragma unroll
    for (int c = 0; c < 4; ++c) {
      st0 = wmma_bf16(ak[2 * c],     Bq[c], st0);
      st1 = wmma_bf16(ak[2 * c + 1], Bq[c], st1);
    }

    const bool need_mask = (kb + 31 > q0);
    float sv0[8], sv1[8];
#pragma unroll
    for (int j = 0; j < 8; ++j) {
      float a = st0[j] * SCL;
      float b = st1[j] * SCL;
      if (need_mask) {
        const int kk = kb + kloc + j;
        if (kk > q0 + hq)      a = -1e30f;
        if (kk + 16 > q0 + hq) b = -1e30f;
      }
      sv0[j] = a; sv1[j] = b;
    }

    // row max (row == fixed q == fixed lane&15, split across lane^16)
    float cm = -1e30f;
#pragma unroll
    for (int j = 0; j < 8; ++j) cm = fmaxf(cm, fmaxf(sv0[j], sv1[j]));
    cm = fmaxf(cm, __shfl_xor(cm, 16, 32));
    const float mnew  = fmaxf(m, cm);
    const float alpha = exp2f(m - mnew);

    float p0[8], p1[8], rs = 0.0f;
#pragma unroll
    for (int j = 0; j < 8; ++j) {
      p0[j] = exp2f(sv0[j] - mnew);
      p1[j] = exp2f(sv1[j] - mnew);
      rs += p0[j] + p1[j];
    }
    rs += __shfl_xor(rs, 16, 32);
    l = l * alpha + rs;
    m = mnew;
#pragma unroll
    for (int t = 0; t < 8; ++t) accO[t] *= alpha;

    // Build P^T B-fragment (32 k x 16 q) from S^T accumulators via xor-16.
    v16bf bp;
#pragma unroll
    for (int j = 0; j < 8; ++j) {
      const float o0 = __shfl_xor(p0[j], 16, 32);
      const float o1 = __shfl_xor(p1[j], 16, 32);
      bp[j]     = (__bf16)((lane < 16) ? p0[j] : o1);
      bp[8 + j] = (__bf16)((lane < 16) ? o0 : p1[j]);
    }

    // Batch all 8 V^T A-fragment loads, then the 8 accumulation WMMAs.
    v16bf av[8];
#pragma unroll
    for (int dt = 0; dt < 8; ++dt)
      av[dt] = load_A(Vt, SEQ, h * HDIM + dt * 16, kb, lane);
#pragma unroll
    for (int dt = 0; dt < 8; ++dt)
      accO[dt] = wmma_bf16(av[dt], bp, accO[dt]);
  }

  const float invl = 1.0f / l;
#pragma unroll
  for (int dt = 0; dt < 8; ++dt) {
    v8bf ob;
#pragma unroll
    for (int j = 0; j < 8; ++j) ob[j] = (__bf16)(accO[dt][j] * invl);
    __bf16* dst = Ob + (size_t)(q0 + hq) * EMBED + h * HDIM + dt * 16 +
                  ((lane < 16) ? 0 : 8);
    *(v8bf*)dst = ob;
  }
}

// ---------------------------------------------------------------- launcher
extern "C" void kernel_launch(void* const* d_in, const int* in_sizes, int n_in,
                              void* d_out, int out_size, void* d_ws,
                              size_t ws_size, hipStream_t stream) {
  const float* x  = (const float*)d_in[0];
  const float* rc = (const float*)d_in[1];
  const float* rs = (const float*)d_in[2];
  const float* Wq = (const float*)d_in[3];
  const float* Wk = (const float*)d_in[4];
  const float* Wv = (const float*)d_in[5];
  const float* Wo = (const float*)d_in[6];

  __bf16* xb  = (__bf16*)d_ws;      // 9 x 8 MiB bf16 buffers = 72 MiB
  __bf16* wqb = xb  + NELEM;
  __bf16* wkb = wqb + NELEM;
  __bf16* wvb = wkb + NELEM;
  __bf16* wob = wvb + NELEM;
  __bf16* Qr  = wob + NELEM;
  __bf16* Kr  = Qr  + NELEM;
  __bf16* Vt  = Kr  + NELEM;
  __bf16* Ob  = Vt  + NELEM;

  const int cvt_blocks = NELEM / 4 / 256;  // 4096
  cvt_f32_to_bf16<<<cvt_blocks, 256, 0, stream>>>(x,  xb);
  cvt_f32_to_bf16<<<cvt_blocks, 256, 0, stream>>>(Wq, wqb);
  cvt_f32_to_bf16<<<cvt_blocks, 256, 0, stream>>>(Wk, wkb);
  cvt_f32_to_bf16<<<cvt_blocks, 256, 0, stream>>>(Wv, wvb);
  cvt_f32_to_bf16<<<cvt_blocks, 256, 0, stream>>>(Wo, wob);

  dim3 ggrid(EMBED / 64, SEQ / 128);       // (32, 16)
  gemm_xwt<<<ggrid, 256, 0, stream>>>(xb, wqb, Qr, nullptr, 0);
  gemm_xwt<<<ggrid, 256, 0, stream>>>(xb, wkb, Kr, nullptr, 0);
  gemm_xwt<<<ggrid, 256, 0, stream>>>(xb, wvb, Vt, nullptr, 1);

  rotary_inplace<<<SEQ * NHEAD * 64 / 256, 256, 0, stream>>>(Qr, Kr, rc, rs);

  flash_attn<<<dim3(NHEAD, SEQ / 16 / 4), 128, 0, stream>>>(Qr, Kr, Vt, Ob);

  gemm_xwt<<<ggrid, 256, 0, stream>>>(Ob, wob, nullptr, (float*)d_out, 2);
}
